// OctreeAttentionDense_28252294873568
// MI455X (gfx1250) — matrix operand order repaired and checked by reading, not compiled, not run on hardware
//
#include <hip/hip_runtime.h>
#include <hip/hip_bf16.h>
#include <cstdint>

// ---------------- problem constants (match reference) ----------------
#define BB   16
#define MM   1024
#define CNT  768
#define DD   256
#define HH   8
#define HD   32
#define TOPK 16
#define NNODES (BB * CNT)          // 12288
#define RROWS  (BB * MM)           // 16384 dense rows
#define SCALE_F 0.17677669529663687f  // 32^-0.5

typedef _Float16 half8  __attribute__((ext_vector_type(8)));
typedef _Float16 half16 __attribute__((ext_vector_type(16)));
typedef float    float8 __attribute__((ext_vector_type(8)));
typedef int      i32x4  __attribute__((vector_size(16)));

#define WMMA16(A, Bf, C) \
  __builtin_amdgcn_wmma_f32_16x16x32_f16(false, (A), false, (Bf), (short)0, (C), false, false)

// ---- CDNA5 async global->LDS copy (guarded; falls back to plain copy) ----
#if defined(__has_builtin)
#  if __has_builtin(__builtin_amdgcn_global_load_async_to_lds_b128) && \
      __has_builtin(__builtin_amdgcn_s_wait_asynccnt)
#    define ASYNC_LDS 1
#  endif
#endif

typedef __attribute__((address_space(1))) i32x4* gas_i32x4;
typedef __attribute__((address_space(3))) i32x4* las_i32x4;

static __device__ __forceinline__ void async_copy16(const _Float16* g, _Float16* l) {
#ifdef ASYNC_LDS
  __builtin_amdgcn_global_load_async_to_lds_b128(
      (gas_i32x4)(uintptr_t)g, (las_i32x4)(uintptr_t)l, 0, 0);
#else
  *(float4*)l = *(const float4*)g;  // 16B copy fallback
#endif
}
static __device__ __forceinline__ void async_wait0() {
#ifdef ASYNC_LDS
  __builtin_amdgcn_s_wait_asynccnt(0);
#endif
}

// Assemble a 16x32 f16 A-fragment from row-major memory.
// Layout: lanes 0-15 hold K 0-15, lanes 16-31 hold K 16-31 (contiguous).
// Caller passes p = row_base + ((lane<16)?0:16); we read p[0..7] and p[8..15].
static __device__ __forceinline__ half16 make_afrag(const _Float16* p) {
  half8 lo = *(const half8*)p;
  half8 hi = *(const half8*)(p + 8);
  half16 r;
#pragma unroll
  for (int i = 0; i < 8; ++i) { r[i] = lo[i]; r[i + 8] = hi[i]; }
  return r;
}

// ---------------- scatter ragged -> dense (f32 + f16 copy) ----------------
__global__ void k_scatter(const float* __restrict__ feat,
                          float* __restrict__ dense,
                          _Float16* __restrict__ dense_h) {
  size_t t = (size_t)blockIdx.x * 256 + threadIdx.x;  // over RROWS*DD
  int row = (int)(t >> 8);
  int c   = (int)(t & 255);
  int b = row >> 10, m = row & 1023;
  float v = 0.0f;
  if (m < CNT) v = feat[((size_t)(b * CNT + m)) * DD + c];
  dense[t]   = v;
  dense_h[t] = (_Float16)v;
}

// ---------------- pack f32 weight [K,N] into WMMA-B f16 tiles ----------------
// Packed layout: [kt][nt][lane(32)][i(16)], kk = kt*32 + (lane>>4)*16 + i,
// n = nt*16 + (lane&15).
__global__ void k_pack_w(const float* __restrict__ W, _Float16* __restrict__ Wp,
                         int K, int N) {
  int t = blockIdx.x * 256 + threadIdx.x;
  if (t >= K * N) return;
  int i = t & 15, lane = (t >> 4) & 31, tile = t >> 9;
  int ntc = N >> 4;
  int kt = tile / ntc, nt = tile % ntc;
  int kk = kt * 32 + (lane >> 4) * 16 + i;
  int n  = nt * 16 + (lane & 15);
  Wp[t] = (_Float16)W[(size_t)kk * N + n];
}

// ---------------- pack V (f16 row-major [R,256]) into per-(b,h) B tiles ----
// vp index: (((b*H + h)*64 + kt)*2 + nt)*512 + lane*16 + i
__global__ void k_pack_v(const _Float16* __restrict__ v, _Float16* __restrict__ vp) {
  size_t t = (size_t)blockIdx.x * 256 + threadIdx.x;  // over RROWS*256
  int i = (int)(t & 15);
  int lane = (int)((t >> 4) & 31);
  int ptile = (int)(t >> 9);
  int nt = ptile & 1;
  int kt = (ptile >> 1) & 63;
  int h  = (ptile >> 7) & 7;
  int b  = ptile >> 10;
  int j   = kt * 32 + (lane >> 4) * 16 + i;
  int col = h * HD + nt * 16 + (lane & 15);
  vp[t] = v[((size_t)(b * MM + j)) * DD + col];
}

// ---------------- slab WMMA GEMM: 128x64 per block, LDS-staged B ------------
// MODE 0: outH = f16(acc * scale)              (QKV)
// MODE 1: outF = resid + acc + bias            (Wout / W2 residual)
// MODE 2: outH = f16(gelu(acc + bias))         (W1)
template <int MODE>
__global__ void k_gemm(const _Float16* __restrict__ A, int lda,
                       const _Float16* __restrict__ Bp,
                       const float* __restrict__ bias,
                       const float* __restrict__ resid,
                       _Float16* __restrict__ outH,
                       float* __restrict__ outF,
                       int Mrows, int K, int N, float scale) {
  __shared__ __align__(64) _Float16 sB[2][2048];  // double-buffered 4KB B panels
  const int lane = threadIdx.x & 31;
  const int wave = threadIdx.x >> 5;
  const int ntc = N >> 4;
  const int nslabs = N >> 6;
  const int rslab = blockIdx.x / nslabs;
  const int cslab = blockIdx.x % nslabs;
  const int ktiles = K >> 5;

  const int r  = lane & 15;
  const int c0 = (lane < 16) ? 0 : 16;
  const _Float16* arow = A + (size_t)(rslab * 128 + wave * 16 + r) * lda + c0;
  const _Float16* bpanel = Bp + (size_t)(cslab * 4) * 512;
  const size_t bstride = (size_t)ntc * 512;
  const int soff = wave * 256 + lane * 8;  // each wave stages 512B of the panel

  async_copy16(bpanel + soff, &sB[0][soff]);

  float8 acc[4] = {};
  for (int kt = 0; kt < ktiles; ++kt) {
    async_wait0();
    __syncthreads();
    int buf = kt & 1;
    if (kt + 1 < ktiles)
      async_copy16(bpanel + (size_t)(kt + 1) * bstride + soff, &sB[buf ^ 1][soff]);
    half16 afrag = make_afrag(arow + kt * 32);
#pragma unroll
    for (int t = 0; t < 4; ++t) {
      half16 bfrag = *(const half16*)(&sB[buf][t * 512 + lane * 16]);
      acc[t] = WMMA16(afrag, bfrag, acc[t]);
    }
  }

  const int cc = lane & 15;
  const int row0 = rslab * 128 + wave * 16 + ((lane < 16) ? 0 : 8);
#pragma unroll
  for (int t = 0; t < 4; ++t) {
    int col = (cslab * 4 + t) * 16 + cc;
#pragma unroll
    for (int g = 0; g < 8; ++g) {
      size_t rr = (size_t)(row0 + g);
      float v = acc[t][g] * scale;
      if (MODE == 0) {
        outH[rr * N + col] = (_Float16)v;
      } else if (MODE == 1) {
        outF[rr * N + col] = resid[rr * N + col] + v + bias[col];
      } else {
        float tt = v + bias[col];
        float gl = 0.5f * tt * (1.0f + erff(tt * 0.70710678118654752f));
        outH[rr * N + col] = (_Float16)gl;
      }
    }
  }
}

// ---------------- fused attention (scores, softmax, PV, head-sum top-k) ----
// grid = (64 row-tiles, B); 256 threads = 8 wave32; 164 KB dynamic LDS.
__global__ void k_attn(const _Float16* __restrict__ q,
                       const _Float16* __restrict__ k,
                       const _Float16* __restrict__ vp,
                       const int* __restrict__ mask,
                       _Float16* __restrict__ attn_out,
                       int* __restrict__ topk_out) {
  extern __shared__ char smem[];
  float*    s_sc   = (float*)smem;                   // [16][1024] scores / PV partials
  float*    s_ps   = s_sc + 16 * 1024;               // [16][1024] head-summed probs
  _Float16* s_pr   = (_Float16*)(s_ps + 16 * 1024);  // [16][1024] f16 probs
  float*    s_bias = (float*)(s_pr + 16 * 1024);     // [1024] mask bias

  const int b  = blockIdx.y;
  const int rt = blockIdx.x;
  const int lane = threadIdx.x & 31;
  const int wave = threadIdx.x >> 5;
  const int r  = lane & 15;
  const int c0 = (lane < 16) ? 0 : 16;
  const int row = threadIdx.x >> 4;
  const int sub = threadIdx.x & 15;

  for (int i = threadIdx.x; i < 16 * 1024; i += 256) s_ps[i] = 0.0f;
  for (int i = threadIdx.x; i < 1024; i += 256)
    s_bias[i] = mask[b * MM + i] ? 0.0f : -10000.0f;
  __syncthreads();

  for (int h = 0; h < HH; ++h) {
    // ---- S = (q*scale) @ k^T : each wave does 8 contiguous column tiles ----
    const _Float16* qp = q + ((size_t)(b * MM + rt * 16 + r)) * DD + h * HD + c0;
    half16 afrag = make_afrag(qp);
    const _Float16* kbase = k + ((size_t)(b * MM + wave * 128 + (lane & 15))) * DD
                              + h * HD + ((lane < 16) ? 0 : 16);
    const int rbase = (lane < 16) ? 0 : 8;
#pragma unroll
    for (int i = 0; i < 8; ++i) {
      half16 bfrag = *(const half16*)(kbase + (size_t)i * 16 * DD);
      float8 cacc = {};
      cacc = WMMA16(afrag, bfrag, cacc);
      int colb = (wave * 8 + i) * 16 + (lane & 15);
#pragma unroll
      for (int g = 0; g < 8; ++g) s_sc[(rbase + g) * 1024 + colb] = cacc[g];
    }
    __syncthreads();

    // ---- masked softmax over 1024 columns (16 threads per row) ----
    float* srow = s_sc + row * 1024;
    float mmax = -1e30f;
    for (int i = 0; i < 64; ++i) {
      int c = sub * 64 + i;
      float sv = srow[c] + s_bias[c];
      srow[c] = sv;
      mmax = fmaxf(mmax, sv);
    }
#pragma unroll
    for (int off = 1; off < 16; off <<= 1) mmax = fmaxf(mmax, __shfl_xor(mmax, off, 32));
    float ssum = 0.0f;
    for (int i = 0; i < 64; ++i) {
      int c = sub * 64 + i;
      float e = __expf(srow[c] - mmax);
      srow[c] = e;
      ssum += e;
    }
#pragma unroll
    for (int off = 1; off < 16; off <<= 1) ssum += __shfl_xor(ssum, off, 32);
    float inv = 1.0f / ssum;
    _Float16* prow16 = s_pr + row * 1024;
    float*    psrow  = s_ps + row * 1024;
    for (int i = 0; i < 64; ++i) {
      int c = sub * 64 + i;
      float p = srow[c] * inv;
      prow16[c] = (_Float16)p;
      psrow[c] += p;
    }
    __syncthreads();

    // ---- out = P @ V : K-partitioned over all 8 waves (6 K-tiles each) ----
    {
      const int nt = wave >> 2, part = wave & 3;
      const _Float16* vbase = vp + (((size_t)(b * HH + h)) * 128 + nt) * 512
                                 + (size_t)lane * 16;
      float8 oacc = {};
#pragma unroll
      for (int kk = 0; kk < 6; ++kk) {
        int kt = part * 6 + kk;  // 4*6*32 = 768 = CNT keys
        half16 pa = make_afrag(s_pr + r * 1024 + kt * 32 + c0);
        half16 bfrag = *(const half16*)(vbase + (size_t)kt * 1024);
        oacc = WMMA16(pa, bfrag, oacc);
      }
      float* slot = s_sc + wave * 256;  // s_sc free: stash partials
#pragma unroll
      for (int g = 0; g < 8; ++g) slot[g * 32 + lane] = oacc[g];
    }
    __syncthreads();
    if (wave < 2) {  // reduce 4 partials, emit f16 head output
      int nt = wave;
#pragma unroll
      for (int g = 0; g < 8; ++g) {
        int e = g * 32 + lane;
        float s = s_sc[(nt * 4 + 0) * 256 + e] + s_sc[(nt * 4 + 1) * 256 + e] +
                  s_sc[(nt * 4 + 2) * 256 + e] + s_sc[(nt * 4 + 3) * 256 + e];
        size_t rr = (size_t)(b * MM + rt * 16 + ((lane < 16) ? 0 : 8) + g);
        attn_out[rr * DD + h * HD + nt * 16 + (lane & 15)] = (_Float16)s;
      }
    }
    __syncthreads();
  }

  // ---- top-k over head-summed probs (lowest-index tie-break, like jax) ----
  float* prow = s_ps + row * 1024;
  for (int t = 0; t < TOPK; ++t) {
    float bv = -1e30f;
    int   bi = 0x7fffffff;
    for (int i = 0; i < 64; ++i) {
      int c = sub * 64 + i;
      float v = prow[c];
      if (v > bv) { bv = v; bi = c; }
    }
#pragma unroll
    for (int off = 1; off < 16; off <<= 1) {
      float ov = __shfl_xor(bv, off, 32);
      int   oi = __shfl_xor(bi, off, 32);
      if (ov > bv || (ov == bv && oi < bi)) { bv = ov; bi = oi; }
    }
    if (sub == 0) topk_out[((size_t)(b * MM + rt * 16 + row)) * TOPK + t] = bi;
    if (sub == (bi >> 6)) prow[bi] = -1e30f;
    __syncthreads();
  }
}

// ---------------- LayerNorm: xn = f16((x-mu)*rstd*gamma + beta) -------------
__global__ void k_ln(const float* __restrict__ x, const float* __restrict__ gamma,
                     const float* __restrict__ beta, _Float16* __restrict__ xn) {
  int rowi = blockIdx.x * 8 + (threadIdx.x >> 5);
  int lane = threadIdx.x & 31;
  const float* xr = x + (size_t)rowi * DD;
  float vals[8];
  float s = 0.0f;
#pragma unroll
  for (int i = 0; i < 8; ++i) { vals[i] = xr[lane + 32 * i]; s += vals[i]; }
#pragma unroll
  for (int off = 1; off < 32; off <<= 1) s += __shfl_xor(s, off, 32);
  float mu = s * (1.0f / DD);
  float vs = 0.0f;
#pragma unroll
  for (int i = 0; i < 8; ++i) { float d = vals[i] - mu; vs += d * d; }
#pragma unroll
  for (int off = 1; off < 32; off <<= 1) vs += __shfl_xor(vs, off, 32);
  float rstd = rsqrtf(vs * (1.0f / DD) + 1e-5f);
  _Float16* out = xn + (size_t)rowi * DD;
#pragma unroll
  for (int i = 0; i < 8; ++i) {
    int c = lane + 32 * i;
    out[c] = (_Float16)((vals[i] - mu) * rstd * gamma[c] + beta[c]);
  }
}

// ---------------- gather dense -> ragged node list --------------------------
__global__ void k_gather(const float* __restrict__ x, const int* __restrict__ slot,
                         float* __restrict__ out) {
  size_t t = (size_t)blockIdx.x * 256 + threadIdx.x;  // over NNODES*DD
  int n = (int)(t >> 8), c = (int)(t & 255);
  out[t] = x[(size_t)slot[n] * DD + c];
}

// ---------------- host-side launch ------------------------------------------
extern "C" void kernel_launch(void* const* d_in, const int* in_sizes, int n_in,
                              void* d_out, int out_size, void* d_ws, size_t ws_size,
                              hipStream_t stream) {
  const float* features = (const float*)d_in[0];
  const float* Wq   = (const float*)d_in[1];
  const float* Wk   = (const float*)d_in[2];
  const float* Wv   = (const float*)d_in[3];
  const float* Wout = (const float*)d_in[4];
  const float* b_out = (const float*)d_in[5];
  const float* gamma = (const float*)d_in[6];
  const float* beta  = (const float*)d_in[7];
  const float* W1 = (const float*)d_in[8];
  const float* b1 = (const float*)d_in[9];
  const float* W2 = (const float*)d_in[10];
  const float* b2 = (const float*)d_in[11];
  const int* slot = (const int*)d_in[12];
  const int* mask = (const int*)d_in[13];

  float* fout = (float*)d_out;
  int* topk_out = (int*)(fout + (size_t)NNODES * DD);

  char* p = (char*)d_ws;
  auto carve = [&](size_t bytes) {
    void* r = (void*)p;
    p += (bytes + 255) & ~(size_t)255;
    return r;
  };
  const size_t RD = (size_t)RROWS * DD;
  float*    dense   = (float*)   carve(RD * 4);
  _Float16* dense_h = (_Float16*)carve(RD * 2);
  _Float16* qh      = (_Float16*)carve(RD * 2);
  _Float16* kh      = (_Float16*)carve(RD * 2);
  _Float16* vh      = (_Float16*)carve(RD * 2);
  _Float16* vpack   = (_Float16*)carve(RD * 2);
  _Float16* wqp     = (_Float16*)carve((size_t)DD * DD * 2);
  _Float16* wkp     = (_Float16*)carve((size_t)DD * DD * 2);
  _Float16* wvp     = (_Float16*)carve((size_t)DD * DD * 2);
  _Float16* woutp   = (_Float16*)carve((size_t)DD * DD * 2);
  _Float16* w1p     = (_Float16*)carve((size_t)DD * 2 * DD * 2);
  _Float16* w2p     = (_Float16*)carve((size_t)2 * DD * DD * 2);
  _Float16* attn_h  = (_Float16*)carve(RD * 2);
  float*    xbuf    = (float*)   carve(RD * 4);
  _Float16* xn      = (_Float16*)carve(RD * 2);
  _Float16* h1      = (_Float16*)carve((size_t)RROWS * 2 * DD * 2);

  k_scatter<<<(int)(RD / 256), 256, 0, stream>>>(features, dense, dense_h);

  k_pack_w<<<(DD * DD) / 256, 256, 0, stream>>>(Wq, wqp, DD, DD);
  k_pack_w<<<(DD * DD) / 256, 256, 0, stream>>>(Wk, wkp, DD, DD);
  k_pack_w<<<(DD * DD) / 256, 256, 0, stream>>>(Wv, wvp, DD, DD);
  k_pack_w<<<(DD * DD) / 256, 256, 0, stream>>>(Wout, woutp, DD, DD);
  k_pack_w<<<(DD * 2 * DD) / 256, 256, 0, stream>>>(W1, w1p, DD, 2 * DD);
  k_pack_w<<<(2 * DD * DD) / 256, 256, 0, stream>>>(W2, w2p, 2 * DD, DD);

  // GEMM grids: 128-row x 64-col slabs per block
  int g256 = (RROWS / 128) * (DD / 64);       // 512 blocks
  int g512 = (RROWS / 128) * (2 * DD / 64);   // 1024 blocks

  k_gemm<0><<<g256, 256, 0, stream>>>(dense_h, DD, wqp, nullptr, nullptr, qh, nullptr,
                                      RROWS, DD, DD, SCALE_F);
  k_gemm<0><<<g256, 256, 0, stream>>>(dense_h, DD, wkp, nullptr, nullptr, kh, nullptr,
                                      RROWS, DD, DD, 1.0f);
  k_gemm<0><<<g256, 256, 0, stream>>>(dense_h, DD, wvp, nullptr, nullptr, vh, nullptr,
                                      RROWS, DD, DD, 1.0f);

  k_pack_v<<<(int)(RD / 256), 256, 0, stream>>>(vh, vpack);

  size_t smem = 16 * 1024 * 4 * 2 + 16 * 1024 * 2 + 1024 * 4;  // 164 KB
  k_attn<<<dim3(MM / 16, BB), 256, smem, stream>>>(qh, kh, vpack, mask, attn_h, topk_out);

  k_gemm<1><<<g256, 256, 0, stream>>>(attn_h, DD, woutp, b_out, dense, nullptr, xbuf,
                                      RROWS, DD, DD, 1.0f);

  k_ln<<<RROWS / 8, 256, 0, stream>>>(xbuf, gamma, beta, xn);

  k_gemm<2><<<g512, 256, 0, stream>>>(xn, DD, w1p, b1, nullptr, h1, nullptr,
                                      RROWS, DD, 2 * DD, 1.0f);

  k_gemm<1><<<g256, 256, 0, stream>>>(h1, 2 * DD, w2p, b2, xbuf, nullptr, dense,
                                      RROWS, 2 * DD, DD, 1.0f);

  k_gather<<<(NNODES * DD) / 256, 256, 0, stream>>>(dense, slot, fout);
}
